// TimeAwareTransformer_30416958390797
// MI455X (gfx1250) — matrix operand order repaired and checked by reading, compiled, not run on hardware
//
#include <hip/hip_runtime.h>

// ---------------- problem constants (from reference) ----------------
#define BB       8
#define SS       2048
#define IN_DIM   8
#define EMB      512
#define HEADS    8
#define HDIM     64          // EMB / HEADS
#define HALF     256         // EMB / 2
#define NCLS     3

typedef _Float16 v16h __attribute__((ext_vector_type(16)));
typedef float    v8f  __attribute__((ext_vector_type(8)));

// pointer types matching the async-to-LDS builtin signature:
//   void (int4 addrspace(1)*, int4 addrspace(3)*, imm int, imm int)
typedef int i4 __attribute__((ext_vector_type(4)));
typedef __attribute__((address_space(1))) i4 gbl_i4;   // global int4
typedef __attribute__((address_space(3))) i4 lds_i4;   // LDS int4

#if defined(__HIP_DEVICE_COMPILE__) && __has_builtin(__builtin_amdgcn_global_load_async_to_lds_b128)
#define USE_ASYNC_LDS 1
#else
#define USE_ASYNC_LDS 0
#endif

__device__ __forceinline__ void wait_async0() {
#if USE_ASYNC_LDS
#if __has_builtin(__builtin_amdgcn_s_wait_asynccnt)
    __builtin_amdgcn_s_wait_asynccnt(0);
#else
    asm volatile("s_wait_asynccnt 0" ::: "memory");
#endif
#endif
}

// =====================================================================
// Kernel 0: convert Wk,Wv (f32) -> f16 once, so GEMM staging is pure DMA
// =====================================================================
__global__ void __launch_bounds__(256)
wcvt_kernel(const float* __restrict__ Wk, const float* __restrict__ Wv,
            _Float16* __restrict__ w16) {
    const int i = blockIdx.x * 256 + threadIdx.x;        // 0 .. 2*EMB*EMB-1
    const float* src = (i < EMB * EMB) ? Wk : Wv;
    w16[i] = (_Float16)src[i & (EMB * EMB - 1)];
}

// =====================================================================
// Kernel 1: x = features@Wf + bf + concat(t*Wl+bl, sin(t*Wp+bp))  -> f16
// =====================================================================
__global__ void __launch_bounds__(256)
x_kernel(const float* __restrict__ feat, const float* __restrict__ ts,
         const float* __restrict__ Wf, const float* __restrict__ bf,
         const float* __restrict__ Wl, const float* __restrict__ bl,
         const float* __restrict__ Wp, const float* __restrict__ bp,
         _Float16* __restrict__ x16) {
    const int token = blockIdx.x;            // 0 .. B*S-1
    const int tid   = threadIdx.x;
    __shared__ float f[IN_DIM];
    if (tid < IN_DIM) f[tid] = feat[(size_t)token * IN_DIM + tid];
    __syncthreads();
    const float t = ts[token];
    for (int e = tid; e < EMB; e += 256) {
        float s = bf[e];
#pragma unroll
        for (int i = 0; i < IN_DIM; ++i) s += f[i] * Wf[i * EMB + e];
        if (e < HALF) s += t * Wl[e] + bl[e];
        else          s += __sinf(t * Wp[e - HALF] + bp[e - HALF]);
        x16[(size_t)token * EMB + e] = (_Float16)s;
    }
}

// =====================================================================
// Kernel 2: K = x@Wk + bk, V = x@Wv + bv  via WMMA f16 (f32 accumulate)
// grid (M/64, N/64, 2), block = 128 threads (4 waves).
// Double-buffered LDS tiles filled with GLOBAL_LOAD_ASYNC_TO_LDS_B128
// (ASYNCcnt pipeline), overlapping next-tile DMA with current WMMAs.
// =====================================================================
#define TM 64
#define TN 64
#define TK 32
#define NKSTEP (EMB / TK)     // 16

__global__ void __launch_bounds__(128)
proj_kv_kernel(const _Float16* __restrict__ x16,
               const _Float16* __restrict__ w16all,   // [2][EMB][EMB] f16 (K then V)
               const float* __restrict__ bk, const float* __restrict__ bv,
               _Float16* __restrict__ kout, _Float16* __restrict__ vout) {
    const _Float16* w16  = w16all + (size_t)blockIdx.z * EMB * EMB;
    const float*    bias = (blockIdx.z == 0) ? bk : bv;
    _Float16*       out  = (blockIdx.z == 0) ? kout : vout;

    __shared__ __align__(16) _Float16 As[2][TM][TK + 8];   // 2 x 64 x 40
    __shared__ __align__(16) _Float16 Bs[2][TK][TN + 8];   // 2 x 32 x 72

    const int tid  = threadIdx.x;
    const int wave = tid >> 5;
    const int lane = tid & 31;
    const int hlf  = lane >> 4;           // lane half (0/1)
    const int mrow = lane & 15;

    const int rowBase = blockIdx.x * TM;
    const int colBase = blockIdx.y * TN;

    // per-thread staging coordinates (16 contiguous f16 = 2 x b128 each)
    const int ar = tid >> 1, ac = (tid & 1) * 16;   // A: 64 rows x 32 cols
    const int br = tid >> 2, bc = (tid & 3) * 16;   // B: 32 rows x 64 cols

    const _Float16* aSrcBase = x16 + (size_t)(rowBase + ar) * EMB + ac;
    const _Float16* bSrcBase = w16 + (size_t)br * EMB + colBase + bc;

    auto stage = [&](int buf, int kk) {
        const _Float16* asrc = aSrcBase + kk;
        const _Float16* bsrc = bSrcBase + (size_t)kk * EMB;
#if USE_ASYNC_LDS
        __builtin_amdgcn_global_load_async_to_lds_b128(
            (gbl_i4*)asrc,       (lds_i4*)&As[buf][ar][ac],     0, 0);
        __builtin_amdgcn_global_load_async_to_lds_b128(
            (gbl_i4*)(asrc + 8), (lds_i4*)&As[buf][ar][ac + 8], 0, 0);
        __builtin_amdgcn_global_load_async_to_lds_b128(
            (gbl_i4*)bsrc,       (lds_i4*)&Bs[buf][br][bc],     0, 0);
        __builtin_amdgcn_global_load_async_to_lds_b128(
            (gbl_i4*)(bsrc + 8), (lds_i4*)&Bs[buf][br][bc + 8], 0, 0);
#else
#pragma unroll
        for (int i = 0; i < 16; ++i) {
            As[buf][ar][ac + i] = asrc[i];
            Bs[buf][br][bc + i] = bsrc[i];
        }
#endif
    };

    v8f acc0 = {}, acc1 = {}, acc2 = {}, acc3 = {};

    stage(0, 0);                               // prologue: fill buffer 0
    for (int i = 0; i < NKSTEP; ++i) {
        const int cur = i & 1;
        wait_async0();                         // my async copies for 'cur' done
        __syncthreads();                       // everyone's copies for 'cur' done
        if (i + 1 < NKSTEP) stage(cur ^ 1, (i + 1) * TK);   // overlap next DMA

        // ---- A fragment: 16x32 f16 (ISA 7.12.2): lane half selects K group ----
        v16h a;
#pragma unroll
        for (int j = 0; j < 8; ++j) {
            const int kb = (j < 4) ? (hlf * 8 + 2 * j)
                                   : (16 + hlf * 8 + 2 * (j - 4));
            a[2 * j]     = As[cur][wave * 16 + mrow][kb];
            a[2 * j + 1] = As[cur][wave * 16 + mrow][kb + 1];
        }
        // ---- B fragments: 32x16 f16, lane&15 = N, element = K (same order as A) ----
        v16h b0, b1f, b2f, b3f;
#pragma unroll
        for (int e = 0; e < 16; ++e) {
            const int kr = (e < 8) ? (hlf * 8 + e) : (16 + hlf * 8 + (e - 8));
            b0[e]  = Bs[cur][kr][ 0 + mrow];
            b1f[e] = Bs[cur][kr][16 + mrow];
            b2f[e] = Bs[cur][kr][32 + mrow];
            b3f[e] = Bs[cur][kr][48 + mrow];
        }
        acc0 = __builtin_amdgcn_wmma_f32_16x16x32_f16(false, a, false, b0,  (short)0, acc0, false, false);
        acc1 = __builtin_amdgcn_wmma_f32_16x16x32_f16(false, a, false, b1f, (short)0, acc1, false, false);
        acc2 = __builtin_amdgcn_wmma_f32_16x16x32_f16(false, a, false, b2f, (short)0, acc2, false, false);
        acc3 = __builtin_amdgcn_wmma_f32_16x16x32_f16(false, a, false, b3f, (short)0, acc3, false, false);
    }

    // ---- store with bias; C layout: VGPR r -> row r + 8*half, col = lane&15 ----
#pragma unroll
    for (int r = 0; r < 8; ++r) {
        const int grow = rowBase + wave * 16 + hlf * 8 + r;
        const int gc   = colBase + (lane & 15);
        _Float16* dst  = out + (size_t)grow * EMB + gc;
        dst[ 0] = (_Float16)(acc0[r] + bias[gc +  0]);
        dst[16] = (_Float16)(acc1[r] + bias[gc + 16]);
        dst[32] = (_Float16)(acc2[r] + bias[gc + 32]);
        dst[48] = (_Float16)(acc3[r] + bias[gc + 48]);
    }
}

// =====================================================================
// Kernel 3: Q for last token only. grid = B, block = 512
// =====================================================================
__global__ void __launch_bounds__(512)
qlast_kernel(const _Float16* __restrict__ x16, const float* __restrict__ Wq,
             const float* __restrict__ bq, float* __restrict__ q) {
    const int b = blockIdx.x, e = threadIdx.x;
    __shared__ float xs[EMB];
    const _Float16* xp = x16 + (size_t)(b * SS + (SS - 1)) * EMB;
    xs[e] = (float)xp[e];
    __syncthreads();
    float s = bq[e];
    for (int k = 0; k < EMB; ++k) s += xs[k] * Wq[(size_t)k * EMB + e];
    q[b * EMB + e] = s;
}

// =====================================================================
// Kernel 4: single-query decayed attention per (b,h). grid = B*HEADS.
// =====================================================================
__global__ void __launch_bounds__(256)
attn_kernel(const float* __restrict__ q, const _Float16* __restrict__ k16,
            const _Float16* __restrict__ v16, const float* __restrict__ ts,
            const float* __restrict__ time_decay, float* __restrict__ out) {
    __shared__ float sc[SS];          // 8 KB of scores
    __shared__ float qs[HDIM];
    __shared__ float red[256];
    __shared__ float oacc[4][HDIM];

    const int b   = blockIdx.x / HEADS;
    const int h   = blockIdx.x % HEADS;
    const int tid = threadIdx.x;

    if (tid < HDIM) qs[tid] = q[b * EMB + h * HDIM + tid];
    const float tL = ts[(size_t)b * SS + (SS - 1)];
    const float td = time_decay[0];
    __syncthreads();

    float lmax = -1e30f;
    for (int t = tid; t < SS; t += 256) {
        const _Float16* kp = k16 + ((size_t)(b * SS + t) * EMB + h * HDIM);
        float s = 0.f;
#pragma unroll
        for (int d = 0; d < HDIM; ++d) s += qs[d] * (float)kp[d];
        s *= 0.125f;                                   // 1/sqrt(HDIM)
        s *= __expf(-td * fabsf(tL - ts[(size_t)b * SS + t]));
        sc[t] = s;
        lmax  = fmaxf(lmax, s);
    }
    red[tid] = lmax; __syncthreads();
    for (int o = 128; o > 0; o >>= 1) {
        if (tid < o) red[tid] = fmaxf(red[tid], red[tid + o]);
        __syncthreads();
    }
    const float m = red[0]; __syncthreads();

    float lsum = 0.f;
    for (int t = tid; t < SS; t += 256) {
        const float p = __expf(sc[t] - m);
        sc[t] = p;
        lsum += p;
    }
    red[tid] = lsum; __syncthreads();
    for (int o = 128; o > 0; o >>= 1) {
        if (tid < o) red[tid] += red[tid + o];
        __syncthreads();
    }
    const float invsum = 1.0f / red[0];

    const int g = tid >> 6;      // 4 groups split the t range
    const int d = tid & 63;
    float acc = 0.f;
    for (int t = g; t < SS; t += 4)
        acc += sc[t] * (float)v16[(size_t)(b * SS + t) * EMB + h * HDIM + d];
    oacc[g][d] = acc; __syncthreads();
    if (g == 0)
        out[b * EMB + h * HDIM + d] =
            (oacc[0][d] + oacc[1][d] + oacc[2][d] + oacc[3][d]) * invsum;
}

// =====================================================================
// Kernel 5: out@Wo+bo -> relu(@W1+b1) -> @W2+b2. grid = B, block = 256.
// =====================================================================
__global__ void __launch_bounds__(256)
head_kernel(const float* __restrict__ a, const float* __restrict__ Wo,
            const float* __restrict__ bo, const float* __restrict__ W1,
            const float* __restrict__ b1, const float* __restrict__ W2,
            const float* __restrict__ b2, float* __restrict__ logits) {
    __shared__ float av[EMB], ov[EMB], hv[HALF];
    const int b = blockIdx.x, tid = threadIdx.x;
    av[tid]       = a[b * EMB + tid];
    av[tid + 256] = a[b * EMB + tid + 256];
    __syncthreads();
    for (int j = tid; j < EMB; j += 256) {
        float s = bo[j];
        for (int k = 0; k < EMB; ++k) s += av[k] * Wo[(size_t)k * EMB + j];
        ov[j] = s;
    }
    __syncthreads();
    {
        float s = b1[tid];
        for (int k = 0; k < EMB; ++k) s += ov[k] * W1[(size_t)k * HALF + tid];
        hv[tid] = fmaxf(s, 0.f);
    }
    __syncthreads();
    if (tid < NCLS) {
        float s = b2[tid];
        for (int k = 0; k < HALF; ++k) s += hv[k] * W2[k * NCLS + tid];
        logits[b * NCLS + tid] = s;
    }
}

// =====================================================================
// host-side launcher
// =====================================================================
extern "C" void kernel_launch(void* const* d_in, const int* in_sizes, int n_in,
                              void* d_out, int out_size, void* d_ws, size_t ws_size,
                              hipStream_t stream) {
    const float* features   = (const float*)d_in[0];
    const float* timestamps = (const float*)d_in[1];
    const float* Wf = (const float*)d_in[2];  const float* bf = (const float*)d_in[3];
    const float* Wl = (const float*)d_in[4];  const float* bl = (const float*)d_in[5];
    const float* Wp = (const float*)d_in[6];  const float* bp = (const float*)d_in[7];
    const float* Wq = (const float*)d_in[8];  const float* bq = (const float*)d_in[9];
    const float* Wk = (const float*)d_in[10]; const float* bk = (const float*)d_in[11];
    const float* Wv = (const float*)d_in[12]; const float* bv = (const float*)d_in[13];
    const float* Wo = (const float*)d_in[14]; const float* bo = (const float*)d_in[15];
    const float* time_decay = (const float*)d_in[16];
    const float* W1 = (const float*)d_in[17]; const float* b1 = (const float*)d_in[18];
    const float* W2 = (const float*)d_in[19]; const float* b2 = (const float*)d_in[20];
    float* logits = (float*)d_out;

    // workspace layout
    const size_t TOK     = (size_t)BB * SS;                    // 16384 tokens
    const size_t X_BYTES = TOK * EMB * sizeof(_Float16);       // 16 MB
    char* w = (char*)d_ws;
    _Float16* x16 = (_Float16*)(w);
    _Float16* k16 = (_Float16*)(w + X_BYTES);
    _Float16* v16 = (_Float16*)(w + 2 * X_BYTES);
    float*    qv  = (float*)(w + 3 * X_BYTES);                           // B*EMB f32
    float*    ao  = (float*)(w + 3 * X_BYTES + BB * EMB * sizeof(float));
    _Float16* w16 = (_Float16*)(w + 3 * X_BYTES + 2 * BB * EMB * sizeof(float));  // 1 MB

    // 0) weights -> f16 (enables pure-DMA staging in the GEMM)
    wcvt_kernel<<<dim3(2 * EMB * EMB / 256), 256, 0, stream>>>(Wk, Wv, w16);

    // 1) token embeddings -> f16
    x_kernel<<<dim3((unsigned)TOK), 256, 0, stream>>>(
        features, timestamps, Wf, bf, Wl, bl, Wp, bp, x16);

    // 2) K and V projections (WMMA GEMM with async-LDS double buffering)
    proj_kv_kernel<<<dim3(TOK / TM, EMB / TN, 2), 128, 0, stream>>>(
        x16, w16, bk, bv, k16, v16);

    // 3) Q for last tokens only
    qlast_kernel<<<dim3(BB), EMB, 0, stream>>>(x16, Wq, bq, qv);

    // 4) decayed single-query attention
    attn_kernel<<<dim3(BB * HEADS), 256, 0, stream>>>(
        qv, k16, v16, timestamps, time_decay, ao);

    // 5) output projection + MLP head -> logits
    head_kernel<<<dim3(BB), 256, 0, stream>>>(ao, Wo, bo, W1, b1, W2, b2, logits);

    (void)in_sizes; (void)n_in; (void)out_size; (void)ws_size;
}
